// StyleBlock_45509473469070
// MI455X (gfx1250) — matrix operand order repaired and checked by reading, not compiled, not run on hardware
//
#include <hip/hip_runtime.h>
#include <stdint.h>

typedef __bf16 v16bf __attribute__((ext_vector_type(16)));
typedef float  v8f   __attribute__((ext_vector_type(8)));
typedef unsigned short v8u  __attribute__((ext_vector_type(8)));
typedef unsigned short v16u __attribute__((ext_vector_type(16)));

#define NB   16
#define CIN  512
#define COUT 512
#define HW   4096
#define LIN_COEF  0.04419417382415922f   /* 1/sqrt(512)  */
#define CONV_COEF 0.01473139127471974f   /* 1/sqrt(4608) */

union U16x16 { v16u u; v16bf b; };

__device__ __forceinline__ unsigned short f2bf(float f) {
    unsigned int u = __float_as_uint(f);
    u += 0x7FFFu + ((u >> 16) & 1u);     // round-to-nearest-even
    return (unsigned short)(u >> 16);
}

// ---------------- prep kernels (bandwidth-trivial) ----------------

// style[b,i] = (w[b,:] . style_w[i,:]) * lin_coef + style_b[i]
__global__ __launch_bounds__(256) void k_style(const float* __restrict__ w,
                                               const float* __restrict__ sw,
                                               const float* __restrict__ sb,
                                               float* __restrict__ style) {
    int t = blockIdx.x * 256 + threadIdx.x;
    if (t >= NB * CIN) return;
    int b = t >> 9, i = t & 511;
    float s = 0.f;
    for (int d = 0; d < 512; ++d) s += w[b * 512 + d] * sw[i * 512 + d];
    style[t] = s * LIN_COEF + sb[i];
}

// s2[o,i] = sum_kk conv_w[o,i,kk]^2
__global__ __launch_bounds__(256) void k_s2(const float* __restrict__ cw,
                                            float* __restrict__ s2) {
    int t = blockIdx.x * 256 + threadIdx.x;
    if (t >= COUT * CIN) return;
    const float* p = cw + (size_t)t * 9;
    float s = 0.f;
    #pragma unroll
    for (int j = 0; j < 9; ++j) s += p[j] * p[j];
    s2[t] = s;
}

// sigma_inv[b,o] = rsqrt(coef^2 * sum_i s2[o,i]*style[b,i]^2 + eps)
__global__ __launch_bounds__(256) void k_sigma(const float* __restrict__ s2,
                                               const float* __restrict__ style,
                                               float* __restrict__ sigma_inv) {
    int t = blockIdx.x * 256 + threadIdx.x;
    if (t >= NB * COUT) return;
    int b = t >> 9, o = t & 511;
    float s = 0.f;
    for (int i = 0; i < 512; ++i) {
        float st = style[b * 512 + i];
        s += s2[o * 512 + i] * st * st;
    }
    sigma_inv[t] = rsqrtf(s * (CONV_COEF * CONV_COEF) + 1e-8f);
}

// xs_t[b][p][i] = bf16( x[b][i][p] * style[b,i] )  -- modulation folded in,
// stored position-major so the conv's im2col gather is wide & contiguous.
// 32x32 LDS-tiled transpose: both global sides coalesced.
__global__ __launch_bounds__(256)
void k_transpose(const float* __restrict__ x, const float* __restrict__ style,
                 unsigned short* __restrict__ xs_t) {
    __shared__ float tile[32][33];
    const int tx = threadIdx.x & 31;
    const int ty = threadIdx.x >> 5;          // 0..7
    const int p0 = blockIdx.x * 32;
    const int i0 = blockIdx.y * 32;
    const int b  = blockIdx.z;
    #pragma unroll
    for (int r = 0; r < 4; ++r) {
        int i = i0 + ty + r * 8;
        tile[ty + r * 8][tx] =
            x[((size_t)b * CIN + i) * HW + p0 + tx] * style[b * CIN + i];
    }
    __syncthreads();
    #pragma unroll
    for (int r = 0; r < 4; ++r) {
        int p = p0 + ty + r * 8;
        xs_t[((size_t)b * HW + p) * CIN + i0 + tx] = f2bf(tile[tx][ty + r * 8]);
    }
}

// Wt9[kk][o][i] = bf16( conv_w[o,i,kk] * conv_coef )  -- tap-major weight layout
__global__ __launch_bounds__(256) void k_w9(const float* __restrict__ cw,
                                            unsigned short* __restrict__ Wt9) {
    int t = blockIdx.x * 256 + threadIdx.x;
    if (t >= 9 * COUT * CIN) return;
    int kk = t / (COUT * CIN);
    int oi = t % (COUT * CIN);
    Wt9[t] = f2bf(cw[(size_t)oi * 9 + kk] * CONV_COEF);
}

// ---------------- main implicit-GEMM conv (bf16 WMMA) ----------------
// grid: (4 O-tiles, 32 P-tiles, 16 batches); block 256 = 8 wave32
// tile: 128(O) x 128(P); K = 9 taps x 512 channels = 144 steps of 32,
// LDS double-buffered, global loads pipelined against WMMA.
__global__ __launch_bounds__(256)
void k_conv(const unsigned short* __restrict__ Wt9,   // [9][512][512] bf16 bits
            const unsigned short* __restrict__ xs_t,  // [B][4096][512] bf16 bits
            const float* __restrict__ sigma_inv,      // [B][512]
            const float* __restrict__ noise,          // [B][4096]
            const float* __restrict__ scale_noise,    // [1]
            const float* __restrict__ bias,           // [512]
            float* __restrict__ out)                  // [B][512][4096]
{
    __shared__ unsigned short As[2][128 * 48];  // 128 O-rows x 32 K (stride 48)
    __shared__ unsigned short Bs[2][128 * 48];  // 128 P-rows x 32 K (transposed im2col)

    const int tid    = threadIdx.x;
    const int lane   = tid & 31;
    const int wid    = tid >> 5;
    const int wave_m = wid >> 2;      // 0..1 -> 64 O-rows
    const int wave_n = wid & 3;       // 0..3 -> 32 P-cols
    const int khalf  = lane >> 4;
    const int lm     = lane & 15;

    const int bm = blockIdx.x;        // O tile 0..3
    const int pn = blockIdx.y;        // spatial tile 0..31 (2 image rows)
    const int b  = blockIdx.z;
    const int y0 = pn * 2;

    // loader mappings: one row / position per thread-pair, 16 bf16 per thread
    const int arow = tid >> 1;
    const int acol = (tid & 1) * 16;
    const int bp   = tid >> 1;                  // tile position 0..127
    const int bc   = (tid & 1) * 16;            // channel chunk
    const int by = y0 + (bp >> 6);
    const int bx = bp & 63;

    const unsigned short* wbase = Wt9 + (size_t)(bm * 128 + arow) * 512 + acol;
    const size_t xplane = (size_t)b * HW;

    v8f acc[4][2];
    #pragma unroll
    for (int i = 0; i < 4; ++i)
        #pragma unroll
        for (int j = 0; j < 2; ++j) acc[i][j] = 0.0f;

    v8u a0, a1, b0, b1;

    auto loadT = [&](int kt) {
        const int kk = kt >> 4;                 // tap (uniform -> SALU)
        const int i0 = (kt & 15) << 5;          // channel block
        const v8u* sA = (const v8u*)(wbase + (size_t)kk * (512 * 512) + i0);
        a0 = sA[0];
        a1 = sA[1];
        const int ys = by + kk / 3 - 1;
        const int xx = bx + kk % 3 - 1;
        v8u z = {0, 0, 0, 0, 0, 0, 0, 0};
        b0 = z;
        b1 = z;
        if (((unsigned)ys < 64u) && ((unsigned)xx < 64u)) {
            const v8u* sB = (const v8u*)(xs_t +
                ((xplane + (size_t)(ys * 64 + xx)) << 9) + i0 + bc);
            b0 = sB[0];                          // two exec-masked b128 loads
            b1 = sB[1];
        }
    };
    auto storeT = [&](int pr) {
        unsigned short* dA = &As[pr][arow * 48 + acol];
        *(v8u*)dA       = a0;
        *(v8u*)(dA + 8) = a1;
        unsigned short* dB = &Bs[pr][bp * 48 + bc];
        *(v8u*)dB       = b0;
        *(v8u*)(dB + 8) = b1;
    };

    loadT(0);
    storeT(0);
    __syncthreads();

    for (int kt = 0; kt < 144; ++kt) {
        const int cur = kt & 1;
        if (kt + 1 < 144) loadT(kt + 1);        // pipeline next tile into regs

        // fragments per ISA 16-bit A/B VGPR layouts (2x ds_load_b128 each)
        v16bf afr[4], bfr[2];
        #pragma unroll
        for (int ms = 0; ms < 4; ++ms) {
            const int row = wave_m * 64 + ms * 16 + lm;
            v8u lo = *(const v8u*)&As[cur][row * 48 + khalf * 8];
            v8u hi = *(const v8u*)&As[cur][row * 48 + 16 + khalf * 8];
            U16x16 cv;
            cv.u = __builtin_shufflevector(lo, hi, 0,1,2,3,4,5,6,7,
                                                   8,9,10,11,12,13,14,15);
            afr[ms] = cv.b;
        }
        #pragma unroll
        for (int ns = 0; ns < 2; ++ns) {
            const int col = wave_n * 32 + ns * 16 + lm;
            v8u lo = *(const v8u*)&Bs[cur][col * 48 + khalf * 8];
            v8u hi = *(const v8u*)&Bs[cur][col * 48 + 16 + khalf * 8];
            U16x16 cv;
            cv.u = __builtin_shufflevector(lo, hi, 0,1,2,3,4,5,6,7,
                                                   8,9,10,11,12,13,14,15);
            bfr[ns] = cv.b;
        }
        #pragma unroll
        for (int ms = 0; ms < 4; ++ms)
            #pragma unroll
            for (int ns = 0; ns < 2; ++ns)
                acc[ms][ns] = __builtin_amdgcn_wmma_f32_16x16x32_bf16(
                    false, afr[ms], false, bfr[ns],
                    (short)0, acc[ms][ns], false, false);

        __syncthreads();                        // all reads of buf[cur^1] done
        if (kt + 1 < 144) storeT(cur ^ 1);
        __syncthreads();                        // buf[cur^1] ready
    }

    // epilogue: demodulate (row scale), + noise, + bias, leaky-ReLU(0.2)
    const float sn = scale_noise[0];
    #pragma unroll
    for (int ms = 0; ms < 4; ++ms) {
        #pragma unroll
        for (int ns = 0; ns < 2; ++ns) {
            const int p = pn * 128 + wave_n * 32 + ns * 16 + lm;
            const float nz = sn * noise[b * HW + p];
            #pragma unroll
            for (int r = 0; r < 8; ++r) {
                const int o = bm * 128 + wave_m * 64 + ms * 16 + khalf * 8 + r;
                float v = acc[ms][ns][r] * sigma_inv[b * 512 + o] + nz + bias[o];
                out[((size_t)b * 512 + o) * HW + p] = (v > 0.f) ? v : 0.2f * v;
            }
        }
    }
}

// ---------------- launch ----------------

extern "C" void kernel_launch(void* const* d_in, const int* in_sizes, int n_in,
                              void* d_out, int out_size, void* d_ws, size_t ws_size,
                              hipStream_t stream) {
    const float* x      = (const float*)d_in[0];
    const float* w      = (const float*)d_in[1];
    const float* noise  = (const float*)d_in[2];
    const float* sw     = (const float*)d_in[3];
    const float* sb     = (const float*)d_in[4];
    const float* cw     = (const float*)d_in[5];
    const float* snoise = (const float*)d_in[6];
    const float* bias   = (const float*)d_in[7];
    float* out = (float*)d_out;

    char* ws = (char*)d_ws;
    float* style          = (float*)(ws + 0);                 //  32 KB
    float* sigma          = (float*)(ws + 32768);             //  32 KB
    float* s2             = (float*)(ws + 65536);             //   1 MB
    unsigned short* Wt9   = (unsigned short*)(ws + 1114112);  // 4.5 MB bf16
    unsigned short* xs_t  = (unsigned short*)(ws + 5832704);  //  64 MB bf16

    k_style<<<(NB * CIN + 255) / 256, 256, 0, stream>>>(w, sw, sb, style);
    k_s2<<<(COUT * CIN + 255) / 256, 256, 0, stream>>>(cw, s2);
    k_sigma<<<(NB * COUT + 255) / 256, 256, 0, stream>>>(s2, style, sigma);
    {
        dim3 g(HW / 32, CIN / 32, NB);
        k_transpose<<<g, 256, 0, stream>>>(x, style, xs_t);
    }
    k_w9<<<(9 * COUT * CIN + 255) / 256, 256, 0, stream>>>(cw, Wt9);

    dim3 grid(4, 32, NB);
    k_conv<<<grid, 256, 0, stream>>>(Wt9, xs_t, sigma, noise, snoise, bias, out);
}
